// Linear8bit_3736621548094
// MI455X (gfx1250) — 8-bit matrix operand order repaired and checked by reading, compiled, not run on hardware
//
#include <hip/hip_runtime.h>
#include <hip/hip_fp16.h>


typedef int v8i __attribute__((ext_vector_type(8)));
typedef int v4i __attribute__((ext_vector_type(4)));

#define M_DIM 8192
#define K_DIM 4096
#define N_DIM 4096
#define KT_COUNT (K_DIM / 64)
#define A_TILE_BYTES 1024
#define B_TILE_BYTES 1024

__global__ __launch_bounds__(256) void pack_a_kernel(
    const int* __restrict__ x, unsigned char* __restrict__ pA) {
  unsigned t = blockIdx.x * 256u + threadIdx.x;
  unsigned chunk = t & 3u;
  unsigned lane  = (t >> 2) & 31u;
  unsigned tile  = t >> 7;
  unsigned kt    = tile & (KT_COUNT - 1u);
  unsigned mt    = tile / KT_COUNT;
  unsigned row   = mt * 16u + (lane & 15u);
  unsigned kbase = kt * 64u + (chunk >> 1) * 32u + (chunk & 1u) * 8u + ((lane >> 4) << 4);

  const v4i* src = (const v4i*)(x + (size_t)row * K_DIM + kbase);
  v4i v0 = src[0];
  v4i v1 = src[1];
  unsigned d0 = (v0[0] & 255) | ((v0[1] & 255) << 8) |
                ((v0[2] & 255) << 16) | ((unsigned)(v0[3] & 255) << 24);
  unsigned d1 = (v1[0] & 255) | ((v1[1] & 255) << 8) |
                ((v1[2] & 255) << 16) | ((unsigned)(v1[3] & 255) << 24);
  uint2 packed; packed.x = d0; packed.y = d1;
  ((uint2*)pA)[t] = packed;
}

__global__ __launch_bounds__(256) void pack_b_kernel(
    const int* __restrict__ w, unsigned char* __restrict__ pB) {
  unsigned t = blockIdx.x * 256u + threadIdx.x;
  unsigned half_ = t & 1u;
  unsigned lane  = (t >> 1) & 31u;
  unsigned tile  = t >> 6;
  unsigned kt    = tile & (KT_COUNT - 1u);
  unsigned nt    = tile / KT_COUNT;
  unsigned col   = nt * 16u + (lane & 15u);
  unsigned kstart = kt * 64u + half_ * 32u + ((lane >> 4) << 4);

  const v4i* src = (const v4i*)(w + (size_t)col * K_DIM + kstart);
  v4i a = src[0], b = src[1], c = src[2], d = src[3];
  uint4 packed;
  packed.x = (a[0] & 255) | ((a[1] & 255) << 8) | ((a[2] & 255) << 16) | ((unsigned)(a[3] & 255) << 24);
  packed.y = (b[0] & 255) | ((b[1] & 255) << 8) | ((b[2] & 255) << 16) | ((unsigned)(b[3] & 255) << 24);
  packed.z = (c[0] & 255) | ((c[1] & 255) << 8) | ((c[2] & 255) << 16) | ((unsigned)(c[3] & 255) << 24);
  packed.w = (d[0] & 255) | ((d[1] & 255) << 8) | ((d[2] & 255) << 16) | ((unsigned)(d[3] & 255) << 24);
  ((uint4*)pB)[t] = packed;
}

__device__ inline v8i load_frag(const unsigned char* p) {
  v4i lo = *(const v4i*)p;
  v4i hi = *(const v4i*)(p + 16);
  return __builtin_shufflevector(lo, hi, 0, 1, 2, 3, 4, 5, 6, 7);
}

__global__ __launch_bounds__(256) void gemm_i8_wmma_kernel(
    const unsigned char* __restrict__ pA,
    const unsigned char* __restrict__ pB,
    const __half* __restrict__ bias,
    float* __restrict__ out) {
  const unsigned lane   = threadIdx.x & 31u;
  const unsigned wave   = threadIdx.x >> 5;
  const unsigned wave_m = wave >> 1;
  const unsigned wave_n = wave & 1u;
  const unsigned mt0 = blockIdx.y * 16u + wave_m * 4u;
  const unsigned nt0 = blockIdx.x * 8u  + wave_n * 4u;
  const unsigned laneOff = lane * 32u;

  const unsigned char* aBase = pA + (size_t)mt0 * KT_COUNT * A_TILE_BYTES + laneOff;
  const unsigned char* bBase = pB + (size_t)nt0 * KT_COUNT * B_TILE_BYTES + laneOff;
  const size_t aTileStride = (size_t)KT_COUNT * A_TILE_BYTES;
  const size_t bTileStride = (size_t)KT_COUNT * B_TILE_BYTES;

  v8i acc[4][4];
#pragma unroll
  for (int i = 0; i < 4; ++i)
#pragma unroll
    for (int j = 0; j < 4; ++j)
      acc[i][j] = (v8i)(0);

  for (unsigned kt = 0; kt < KT_COUNT; ++kt) {
    const unsigned char* ap = aBase + (size_t)kt * A_TILE_BYTES;
    const unsigned char* bp = bBase + (size_t)kt * B_TILE_BYTES;

    v8i afrag[4];
#pragma unroll
    for (int i = 0; i < 4; ++i)
      afrag[i] = load_frag(ap + (size_t)i * aTileStride);
    v8i bfrag[4];
#pragma unroll
    for (int j = 0; j < 4; ++j)
      bfrag[j] = load_frag(bp + (size_t)j * bTileStride);

#pragma unroll
    for (int i = 0; i < 4; ++i)
#pragma unroll
      for (int j = 0; j < 4; ++j)
        acc[i][j] = __builtin_amdgcn_wmma_i32_16x16x64_iu8(
             true, afrag[i],  true, bfrag[j],
            acc[i][j],  false,  false);
  }

  const unsigned rowHalf   = (lane >> 4) << 3;
  const unsigned colInTile = lane & 15u;
#pragma unroll
  for (int i = 0; i < 4; ++i) {
    const unsigned rbase = (mt0 + i) * 16u + rowHalf;
#pragma unroll
    for (int j = 0; j < 4; ++j) {
      const unsigned col = (nt0 + j) * 16u + colInTile;
      const __half bv = bias[col];
#pragma unroll
      for (int r = 0; r < 8; ++r) {
        __half h = __hadd(__float2half_rn((float)acc[i][j][r]), bv);
        __builtin_nontemporal_store((float)h, out + (size_t)(rbase + r) * N_DIM + col);
      }
    }
  }
}

extern "C" void kernel_launch(void* const* d_in, const int* in_sizes, int n_in,
                              void* d_out, int out_size, void* d_ws, size_t ws_size,
                              hipStream_t stream) {
  (void)in_sizes; (void)n_in; (void)out_size; (void)ws_size;
  const int* x     = (const int*)d_in[0];
  const int* wmat  = (const int*)d_in[1];
  const __half* bias = (const __half*)d_in[2];
  float* out = (float*)d_out;

  unsigned char* pA = (unsigned char*)d_ws;
  unsigned char* pB = pA + (size_t)M_DIM * K_DIM;

  pack_a_kernel<<<(M_DIM * (size_t)K_DIM / 8) / 256, 256, 0, stream>>>(x, pA);
  pack_b_kernel<<<(N_DIM * (size_t)K_DIM / 16) / 256, 256, 0, stream>>>(wmat, pB);

  dim3 grid(N_DIM / 128, M_DIM / 256);
  gemm_i8_wmma_kernel<<<grid, 256, 0, stream>>>(pA, pB, bias, out);
}
